// Reservoir_78314433675797
// MI455X (gfx1250) — compile-verified, matrix operand order checked by reading, not run on hardware
//
#include <hip/hip_runtime.h>
#include <hip/hip_bf16.h>

typedef __attribute__((ext_vector_type(16))) __bf16 v16bf;
typedef __attribute__((ext_vector_type(8)))  __bf16 v8bf;
typedef __attribute__((ext_vector_type(8)))  float  v8f;

namespace {
constexpr int   kT       = 2048;
constexpr int   kB       = 32;
constexpr int   kIn      = 128;
constexpr int   kH       = 2048;
constexpr float kAlpha   = 0.9f;
constexpr int   kBlocks  = 64;     // persistent blocks, must all be co-resident
constexpr int   kThreads = 128;    // 4 wave32 waves
constexpr int   kCols    = kH / kBlocks;     // 32 columns of H per block
constexpr size_t kStateElems = (size_t)kB * kH;

// workspace byte offsets
constexpr size_t WB_OFF  = 0;                                   // W_hat bf16 [H][H]      (8 MB)
constexpr size_t WIN_OFF = WB_OFF  + (size_t)kH * kH  * 2;      // W_in  bf16 [H][IN]     (512 KB)
constexpr size_t ST_OFF  = WIN_OFF + (size_t)kH * kIn * 2;      // state bf16 ping+pong   (256 KB)
constexpr size_t XB_OFF  = ST_OFF  + 2 * kStateElems * 2;       // x bf16 [T][B][IN]      (16 MB)
constexpr size_t CNT_OFF = XB_OFF  + (size_t)kT * kB * kIn * 2; // barrier counter
} // namespace

// ---------------------------------------------------------------------------
// Init: fp32 -> bf16 conversion of W_hat, W_in and the whole x tensor; seed
// the bf16 state ping buffer from initial_state; zero the barrier counter.
// Re-run every call (deterministic, no caching).
// ---------------------------------------------------------------------------
__global__ void reservoir_init_kernel(const float* __restrict__ W_hat,
                                      const float* __restrict__ W_in,
                                      const float* __restrict__ h0,
                                      const float* __restrict__ x,
                                      __bf16* __restrict__ Wb,
                                      __bf16* __restrict__ Winb,
                                      __bf16* __restrict__ st,
                                      __bf16* __restrict__ xb,
                                      unsigned* __restrict__ counter) {
  const size_t stride = (size_t)gridDim.x * blockDim.x;
  const size_t tid    = (size_t)blockIdx.x * blockDim.x + threadIdx.x;
  const size_t n_wh = (size_t)kH * kH;
  const size_t n_wi = (size_t)kH * kIn;
  const size_t n_x  = (size_t)kT * kB * kIn;
  for (size_t i = tid; i < n_wh;        i += stride) Wb[i]   = (__bf16)W_hat[i];
  for (size_t i = tid; i < n_wi;        i += stride) Winb[i] = (__bf16)W_in[i];
  for (size_t i = tid; i < kStateElems; i += stride) st[i]   = (__bf16)h0[i];
  for (size_t i = tid; i < n_x;         i += stride) xb[i]   = (__bf16)x[i];
  if (tid == 0) *counter = 0u;
}

// ---------------------------------------------------------------------------
// Persistent reservoir scan. Each wave owns one 16x16 output tile of the
// [B=32, H] state for all 2048 timesteps; the leaky carry lives in registers.
// Per step: acc = x_t @ W_in^T (K=128, 4 WMMAs) + state @ W_hat^T (K=2048,
// 64 WMMAs), all v_wmma_f32_16x16x32_bf16 with fp32 accumulation, then
// state = 0.1*prev + 0.9*tanh(acc), written fp32 to out and bf16 to the
// ping-pong broadcast buffer, followed by a device-wide atomic barrier.
// Ping/pong selected by integer offset from ONE base pointer so address-space
// inference keeps every load a global_load (no flat).
// ---------------------------------------------------------------------------
__global__ __launch_bounds__(kThreads)
void reservoir_scan_kernel(const __bf16* __restrict__ xb,     // [T][B][IN] bf16
                           const float* __restrict__ h0,      // [B][H]     fp32
                           const __bf16* __restrict__ Wb,     // [H][H]     bf16
                           const __bf16* __restrict__ Winb,   // [H][IN]    bf16
                           __bf16* __restrict__ st,           // 2x [B][H]  bf16
                           float* __restrict__ out,           // [T][B][H]  fp32
                           unsigned* __restrict__ counter) {
  const int lane  = threadIdx.x & 31;
  const int wave  = threadIdx.x >> 5;     // 0..3
  const int half  = lane >> 4;            // lane group 0/1
  const int l15   = lane & 15;
  const int m0    = (wave & 1) * 16;      // batch-row tile base (0 or 16)
  const int j0    = blockIdx.x * kCols + (wave >> 1) * 16;  // column tile base

  const int arow  = m0 + l15;             // A-fragment source row (batch)
  const int brow  = j0 + l15;             // B-fragment source row (W row = out col)

  // leaky-integrator carry for this wave's 16x16 D tile (fp32, in registers)
  float carry[8];
#pragma unroll
  for (int r = 0; r < 8; ++r) {
    const int b = m0 + r + 8 * half;      // D layout: VGPR r -> M = r (+8 for lanes 16..31)
    carry[r] = h0[(size_t)b * kH + j0 + l15];
  }

  const __bf16* __restrict__ WinRow = Winb + (size_t)brow * kIn;
  const __bf16* __restrict__ WRow   = Wb   + (size_t)brow * kH;
  const size_t  arowOff = (size_t)arow * kH;

  for (int t = 0; t < kT; ++t) {
    // integer-offset ping/pong (keeps loads in addrspace(1) -> global_load)
    const size_t sOff  = (size_t)(t & 1) * kStateElems;
    const size_t snOff = (size_t)((t + 1) & 1) * kStateElems;

    v8f acc = {};

    // ---- fused input projection: A = x_t (bf16), B = W_in^T ----
    const __bf16* __restrict__ xt = xb + ((size_t)t * kB + arow) * kIn + half * 8;
#pragma unroll
    for (int k = 0; k < kIn; k += 32) {
      union { v16bf v; v8bf h[2]; } a, bm;
      a.h[0]  = *(const v8bf*)(xt + k);
      a.h[1]  = *(const v8bf*)(xt + k + 16);
      const __bf16* pb = WinRow + k + half * 8;
      bm.h[0] = *(const v8bf*)(pb);
      bm.h[1] = *(const v8bf*)(pb + 16);
      acc = __builtin_amdgcn_wmma_f32_16x16x32_bf16(false, a.v, false, bm.v,
                                                    (short)0, acc, false, false);
    }

    // ---- recurrent GEMM: A = state (bf16 broadcast), B = W_hat^T ----
    const __bf16* __restrict__ Arow = st + sOff + arowOff + half * 8;
    const __bf16* __restrict__ Brow = WRow + half * 8;
#pragma unroll 4
    for (int k = 0; k < kH; k += 32) {
      union { v16bf v; v8bf h[2]; } a, bm;
      a.h[0]  = *(const v8bf*)(Arow + k);
      a.h[1]  = *(const v8bf*)(Arow + k + 16);
      bm.h[0] = *(const v8bf*)(Brow + k);
      bm.h[1] = *(const v8bf*)(Brow + k + 16);
      acc = __builtin_amdgcn_wmma_f32_16x16x32_bf16(false, a.v, false, bm.v,
                                                    (short)0, acc, false, false);
    }

    // ---- activation + leaky update; write fp32 result + bf16 broadcast ----
    float*  __restrict__ orow = out + (size_t)t * kB * kH;
    __bf16* __restrict__ SN   = st + snOff;
#pragma unroll
    for (int r = 0; r < 8; ++r) {
      const int b = m0 + r + 8 * half;
      const float s = (1.0f - kAlpha) * carry[r] + kAlpha * tanhf(acc[r]);
      carry[r] = s;
      const size_t idx = (size_t)b * kH + j0 + l15;
      orow[idx] = s;
      SN[idx]   = (__bf16)s;
    }

    // ---- device-wide barrier: all blocks publish step t before step t+1 ----
    __syncthreads();
    if (threadIdx.x == 0) {
      __threadfence();  // make bf16 state stores visible at device scope
      __hip_atomic_fetch_add(counter, 1u, __ATOMIC_RELEASE,
                             __HIP_MEMORY_SCOPE_AGENT);
      const unsigned target = (unsigned)gridDim.x * (unsigned)(t + 1);
      while (__hip_atomic_load(counter, __ATOMIC_ACQUIRE,
                               __HIP_MEMORY_SCOPE_AGENT) < target) {
        __builtin_amdgcn_s_sleep(2);
      }
    }
    __syncthreads();
  }
}

extern "C" void kernel_launch(void* const* d_in, const int* in_sizes, int n_in,
                              void* d_out, int out_size, void* d_ws, size_t ws_size,
                              hipStream_t stream) {
  (void)in_sizes; (void)n_in; (void)out_size; (void)ws_size;

  const float* x     = (const float*)d_in[0];   // [T,B,IN]
  const float* h0    = (const float*)d_in[1];   // [B,H]
  const float* W_in  = (const float*)d_in[2];   // [H,IN]
  const float* W_hat = (const float*)d_in[3];   // [H,H]
  float*       out   = (float*)d_out;           // [T,B,H]

  char* ws = (char*)d_ws;
  __bf16*   Wb      = (__bf16*)(ws + WB_OFF);
  __bf16*   Winb    = (__bf16*)(ws + WIN_OFF);
  __bf16*   st      = (__bf16*)(ws + ST_OFF);
  __bf16*   xb      = (__bf16*)(ws + XB_OFF);
  unsigned* counter = (unsigned*)(ws + CNT_OFF);

  reservoir_init_kernel<<<2048, 256, 0, stream>>>(W_hat, W_in, h0, x, Wb, Winb,
                                                  st, xb, counter);
  reservoir_scan_kernel<<<kBlocks, kThreads, 0, stream>>>(xb, h0, Wb, Winb, st,
                                                          out, counter);
}
